// selfattention_59416577573049
// MI455X (gfx1250) — matrix operand order repaired and checked by reading, not compiled, not run on hardware
//
#include <hip/hip_runtime.h>
#include <hip/hip_bf16.h>

typedef __bf16 bf16_t;
typedef __attribute__((ext_vector_type(16))) __bf16 v16bf;
typedef __attribute__((ext_vector_type(2)))  __bf16 bf16x2;
typedef __attribute__((ext_vector_type(8)))  float  v8f;
typedef __attribute__((ext_vector_type(2)))  float  f32x2;

#define B_   4
#define L_   1024
#define D_   1024
#define H_   16
#define HD_  64
#define TD_  3072
#define SCALE_ 0.125f   // 1/sqrt(64)

__device__ __forceinline__ v8f wmma_bf16(v16bf a, v16bf b, v8f c) {
  // D = A(16x32) * B(32x16) + C, fp32 accumulate
  return __builtin_amdgcn_wmma_f32_16x16x32_bf16(false, a, false, b, (short)0, c, false, false);
}

// ---------------------------------------------------------------------------
// Kernel 0: W [1024,3072] fp32 -> Wt [3072,1024] bf16, LDS-tiled transpose so
// both the fp32 reads and the bf16 writes are fully coalesced.
// ---------------------------------------------------------------------------
__global__ __launch_bounds__(256) void wqkv_transpose_kernel(
    const float* __restrict__ W, bf16_t* __restrict__ Wt) {
  __shared__ float tile[32][33];
  const int tx = threadIdx.x & 31;
  const int ty = threadIdx.x >> 5;            // 0..7
  const int n0 = blockIdx.x * 32;             // along TD (96 blocks)
  const int k0 = blockIdx.y * 32;             // along D  (32 blocks)
#pragma unroll
  for (int i = 0; i < 4; ++i)
    tile[ty + i * 8][tx] = W[(size_t)(k0 + ty + i * 8) * TD_ + n0 + tx];
  __syncthreads();
#pragma unroll
  for (int i = 0; i < 4; ++i)
    Wt[(size_t)(n0 + ty + i * 8) * D_ + k0 + tx] = (bf16_t)tile[tx][ty + i * 8];
}

// ---------------------------------------------------------------------------
// Kernel 1: QKV projection. Each wave computes a 16x128 tile of x @ Wqkv + b
// (A fragment reused across 8 WMMAs per k-step), epilogue scatters into
// Qbf/Kbf [B,H,L,HD] and Vt [B,H,HD,L] (all bf16).
// ---------------------------------------------------------------------------
__global__ __launch_bounds__(128) void qkv_gemm_kernel(
    const float* __restrict__ x, const float* __restrict__ bqkv,
    const bf16_t* __restrict__ Wt,
    bf16_t* __restrict__ Qbf, bf16_t* __restrict__ Kbf, bf16_t* __restrict__ Vt) {
  const int lane = threadIdx.x & 31;
  const int wave = threadIdx.x >> 5;
  const int n16  = lane & 15;
  const int hi   = lane >> 4;

  const int tile  = blockIdx.x * 4 + wave;   // (B*L/16) * (3D/128) = 256*24
  const int mtile = tile / 24;
  const int ntile = tile % 24;
  const int m0 = mtile * 16;
  const int n0 = ntile * 128;

  v8f acc[8] = {v8f{}, v8f{}, v8f{}, v8f{}, v8f{}, v8f{}, v8f{}, v8f{}};

  for (int k0 = 0; k0 < D_; k0 += 32) {
    // A fragment: 16 rows of x, K=32 chunk, fp32 -> bf16
    v16bf a;
    const float* xrow = x + (size_t)(m0 + n16) * D_ + k0;
#pragma unroll
    for (int v = 0; v < 8; ++v) {
      const int g = v >> 2, vv = v & 3;
      const int kk = hi * 16 + 2 * v;
      f32x2 p = *(const f32x2*)(xrow + kk);
      a[2 * v]     = (bf16_t)p.x;
      a[2 * v + 1] = (bf16_t)p.y;
    }
#pragma unroll
    for (int c = 0; c < 8; ++c) {
      // B fragment: B[k,n] = W[k0+k, n0+c*16+n16] = Wt[(col)*D + k0+k]
      v16bf bm;
      const bf16_t* wcol = Wt + (size_t)(n0 + c * 16 + n16) * D_ + k0 + hi * 16;
#pragma unroll
      for (int v = 0; v < 8; ++v) {
        bf16x2 p = *(const bf16x2*)(wcol + 2 * v);
        bm[2 * v]     = p.x;
        bm[2 * v + 1] = p.y;
      }
      acc[c] = wmma_bf16(a, bm, acc[c]);
    }
  }

  // Epilogue: bias + scatter. Each 16-col chunk lies inside one (which, head).
#pragma unroll
  for (int c = 0; c < 8; ++c) {
    const int col0  = n0 + c * 16;            // multiple of 16
    const int which = col0 >> 10;             // 0=Q 1=K 2=V (uniform per chunk)
    const int hh    = (col0 & (D_ - 1)) >> 6;
    const int hd    = (col0 & (HD_ - 1)) + n16;
    const float bv  = bqkv[col0 + n16];
#pragma unroll
    for (int v = 0; v < 8; ++v) {
      const int row = m0 + v + 8 * hi;        // global row in [0, B*L)
      const int bb  = row >> 10;
      const int ll  = row & (L_ - 1);
      const float val = acc[c][v] + bv;
      if (which == 2) {
        Vt[((size_t)(bb * H_ + hh) * HD_ + hd) * L_ + ll] = (bf16_t)val;
      } else {
        bf16_t* dst = (which == 0) ? Qbf : Kbf;
        dst[((size_t)(bb * H_ + hh) * L_ + ll) * HD_ + hd] = (bf16_t)val;
      }
    }
  }
}

// ---------------------------------------------------------------------------
// Kernel 2: flash attention with relative_key_query bias.
// One wave owns 16 query rows (l0..l0+15) of one (b,h); loops r in steps of 32.
// Computes S^T = K@Q^T so softmax rows live on lanes and P^T packs into the
// A-fragment of P@V with no cross-lane traffic.
// ---------------------------------------------------------------------------
__global__ __launch_bounds__(128) void attn_rel_kernel(
    const bf16_t* __restrict__ Qbf, const bf16_t* __restrict__ Kbf,
    const bf16_t* __restrict__ Vt, const float* __restrict__ dist_emb,
    float* __restrict__ out) {
  __shared__ float lds_rq[4][16 * 32];       // per-wave rel_q staging tile

  const int lane = threadIdx.x & 31;
  const int wave = threadIdx.x >> 5;
  const int n16  = lane & 15;
  const int hi   = lane >> 4;

  const int tile = blockIdx.x * 4 + wave;    // B*H*(L/16) = 4096 tiles
  const int l0   = (tile & 63) * 16;
  const int bh   = tile >> 6;                // 0..63
  const size_t base = (size_t)bh * L_ * HD_;
  const bf16_t* Qp = Qbf + base;
  const bf16_t* Kp = Kbf + base;
  const bf16_t* Vp = Vt + base;              // [HD][L] for this (b,h)
  float* rqlds = lds_rq[wave];

  // Q loaded once: as B-fragment (Q^T, for S^T) and as A-fragment (for rel_q)
  v16bf bq[2], aq[2];
  {
    const bf16_t* qrow = Qp + (size_t)(l0 + n16) * HD_;
#pragma unroll
    for (int kc = 0; kc < 2; ++kc) {
#pragma unroll
      for (int v = 0; v < 8; ++v) {
        bf16x2 pb = *(const bf16x2*)(qrow + kc * 32 + (v >> 2) * 16 + hi * 8 + 2 * (v & 3));
        bq[kc][2 * v]     = pb.x;
        bq[kc][2 * v + 1] = pb.y;
        const int g = v >> 2, vv = v & 3;
        bf16x2 pa = *(const bf16x2*)(qrow + kc * 32 + g * 16 + hi * 8 + 2 * vv);
        aq[kc][2 * v]     = pa.x;
        aq[kc][2 * v + 1] = pa.y;
      }
    }
  }

  float m_run = -1e30f, l_run = 0.0f;
  v8f O[4] = {v8f{}, v8f{}, v8f{}, v8f{}};

  for (int r0 = 0; r0 < L_; r0 += 32) {
    // prefetch next K tile rows (global_prefetch_b8)
    if (r0 + 32 < L_) {
      __builtin_prefetch(Kp + (size_t)(r0 + 32 + n16) * HD_ + hi * 32, 0, 3);
      __builtin_prefetch(Kp + (size_t)(r0 + 48 + n16) * HD_ + hi * 32, 0, 3);
    }

    float pbuf[2][8];
    float mx = m_run;

#pragma unroll
    for (int sub = 0; sub < 2; ++sub) {
      const int rs = r0 + sub * 16;
      // K tile as A fragments (rows = r)
      v16bf ak[2];
      const bf16_t* krow = Kp + (size_t)(rs + n16) * HD_;
#pragma unroll
      for (int kc = 0; kc < 2; ++kc) {
#pragma unroll
        for (int v = 0; v < 8; ++v) {
          const int g = v >> 2, vv = v & 3;
          bf16x2 p = *(const bf16x2*)(krow + kc * 32 + g * 16 + hi * 8 + 2 * vv);
          ak[kc][2 * v]     = p.x;
          ak[kc][2 * v + 1] = p.y;
        }
      }
      // S^T[r', l'] = sum_d K[rs+r',d] * Q[l0+l',d]
      v8f st = v8f{};
      st = wmma_bf16(ak[0], bq[0], st);
      st = wmma_bf16(ak[1], bq[1], st);

      // dist_emb window: j = l - r + (L-1); jlo so that jcol = l' - r' + 15
      const int jlo = l0 - rs + (L_ - 16);   // always >= 0
      v16bf be[2][2];
#pragma unroll
      for (int tj = 0; tj < 2; ++tj) {
        int j = jlo + tj * 16 + n16;
        j = (j > 2 * L_ - 2) ? (2 * L_ - 2) : j;   // clamp (unused cells only)
        const float* erow = dist_emb + (size_t)j * HD_;
#pragma unroll
        for (int kc = 0; kc < 2; ++kc) {
#pragma unroll
          for (int v = 0; v < 8; ++v) {
            f32x2 p = *(const f32x2*)(erow + kc * 32 + (v >> 2) * 16 + hi * 8 + 2 * (v & 3));
            be[tj][kc][2 * v]     = (bf16_t)p.x;
            be[tj][kc][2 * v + 1] = (bf16_t)p.y;
          }
        }
      }
      // Rk[r', j'] = K[rs+r'] . E[jlo + j']   (same A as S^T)
      v8f rk0 = v8f{}, rk1 = v8f{};
      rk0 = wmma_bf16(ak[0], be[0][0], rk0);
      rk0 = wmma_bf16(ak[1], be[0][1], rk0);
      rk1 = wmma_bf16(ak[0], be[1][0], rk1);
      rk1 = wmma_bf16(ak[1], be[1][1], rk1);
      // Rq[l', j'] = Q[l0+l'] . E[jlo + j']
      v8f rq0 = v8f{}, rq1 = v8f{};
      rq0 = wmma_bf16(aq[0], be[0][0], rq0);
      rq0 = wmma_bf16(aq[1], be[0][1], rq0);
      rq1 = wmma_bf16(aq[0], be[1][0], rq1);
      rq1 = wmma_bf16(aq[1], be[1][1], rq1);

      // stage Rq through LDS (gather needs per-lane row AND column)
      __syncthreads();
#pragma unroll
      for (int v = 0; v < 8; ++v) {
        rqlds[(v + 8 * hi) * 32 + n16]      = rq0[v];
        rqlds[(v + 8 * hi) * 32 + 16 + n16] = rq1[v];
      }
      __syncthreads();

      // combine: s = (S^T + rel_q + rel_k) * scale ; lane col = l', vgpr = r'
#pragma unroll
      for (int v = 0; v < 8; ++v) {
        const int rloc = v + 8 * hi;
        const int jcol = n16 - rloc + 15;          // in [0,30]
        const float rqv = rqlds[n16 * 32 + jcol];
        const int src = (jcol & 15) + (hi << 4);   // Rk row-half == hi
        const float rka = __shfl(rk0[v], src, 32);
        const float rkb = __shfl(rk1[v], src, 32);
        const float rkv = (jcol < 16) ? rka : rkb;
        const float s = (st[v] + rqv + rkv) * SCALE_;
        pbuf[sub][v] = s;
        mx = fmaxf(mx, s);
      }
    }

    // online softmax over this 32-col block (row l' = lane%16)
    mx = fmaxf(mx, __shfl_xor(mx, 16, 32));
    const float sc = __expf(m_run - mx);
    float psum = 0.0f;
#pragma unroll
    for (int sub = 0; sub < 2; ++sub)
#pragma unroll
      for (int v = 0; v < 8; ++v) {
        const float e = __expf(pbuf[sub][v] - mx);
        pbuf[sub][v] = e;
        psum += e;
      }
    psum += __shfl_xor(psum, 16, 32);
    l_run = l_run * sc + psum;
    m_run = mx;

    // rescale O rows (O row l = v + 8*hi; state lives at lane l)
#pragma unroll
    for (int v = 0; v < 8; ++v) {
      const float scv = __shfl(sc, v + 8 * hi, 32);
#pragma unroll
      for (int c = 0; c < 4; ++c) O[c][v] *= scv;
    }

    // Pack P^T accumulator directly into A fragment of P (in-lane remap)
    v16bf ap;
#pragma unroll
    for (int v = 0; v < 8; ++v) {
      const int g = v >> 2, vv = v & 3;
      ap[2 * v]     = (bf16_t)pbuf[g][2 * vv];
      ap[2 * v + 1] = (bf16_t)pbuf[g][2 * vv + 1];
    }

    // O += P(16x32) @ V(32x64), V stored transposed [HD][L]
#pragma unroll
    for (int c = 0; c < 4; ++c) {
      v16bf bv;
      const bf16_t* vrow = Vp + (size_t)(c * 16 + n16) * L_ + r0 + hi * 8;
#pragma unroll
      for (int v = 0; v < 8; ++v) {
        bf16x2 p = *(const bf16x2*)(vrow + (v >> 2) * 16 + 2 * (v & 3));
        bv[2 * v]     = p.x;
        bv[2 * v + 1] = p.y;
      }
      O[c] = wmma_bf16(ap, bv, O[c]);
    }
  }

  // normalize and store: out[b, l, h*64 + hd], fp32
  const float inv = 1.0f / l_run;
  const int b = bh >> 4, h = bh & 15;
#pragma unroll
  for (int v = 0; v < 8; ++v) {
    const float iv = __shfl(inv, v + 8 * hi, 32);
    const int ll = l0 + v + 8 * hi;
    float* orow = out + ((size_t)(b * L_ + ll)) * D_ + h * HD_;
#pragma unroll
    for (int c = 0; c < 4; ++c) orow[c * 16 + n16] = O[c][v] * iv;
  }
}

// ---------------------------------------------------------------------------
extern "C" void kernel_launch(void* const* d_in, const int* in_sizes, int n_in,
                              void* d_out, int out_size, void* d_ws, size_t ws_size,
                              hipStream_t stream) {
  const float* x        = (const float*)d_in[0];   // [B,L,D]
  const float* Wqkv     = (const float*)d_in[1];   // [D,3D]
  const float* bqkv     = (const float*)d_in[2];   // [3D]
  const float* dist_emb = (const float*)d_in[3];   // [2L-1, HD]
  float* out = (float*)d_out;

  char* ws = (char*)d_ws;
  bf16_t* Qbf = (bf16_t*)(ws);                       //  8 MB [B,H,L,HD]
  bf16_t* Kbf = (bf16_t*)(ws + ((size_t)8 << 20));   //  8 MB [B,H,L,HD]
  bf16_t* Vt  = (bf16_t*)(ws + ((size_t)16 << 20));  //  8 MB [B,H,HD,L]
  bf16_t* Wt  = (bf16_t*)(ws + ((size_t)24 << 20));  //  6 MB [3D,D]

  // 0) transpose+convert weights (coalesced both directions via LDS tile)
  wqkv_transpose_kernel<<<dim3(TD_ / 32, D_ / 32), 256, 0, stream>>>(Wqkv, Wt);
  // 1) QKV projection: (B*L/16)*(3D/128) = 6144 waves, 4 waves/block
  qkv_gemm_kernel<<<1536, 128, 0, stream>>>(x, bqkv, Wt, Qbf, Kbf, Vt);
  // 2) flash attention with relative bias: 4096 waves, 4 waves/block
  attn_rel_kernel<<<1024, 128, 0, stream>>>(Qbf, Kbf, Vt, dist_emb, out);
}
